// RaggedAttention_41953240547783
// MI455X (gfx1250) — matrix operand order repaired and checked by reading, not compiled, not run on hardware
//
#include <hip/hip_runtime.h>

// ---------------------------------------------------------------------------
// CDNA5 (gfx1250) ragged attention: gather + QKV + flash attention + proj.
// GEMMs via v_wmma_f32_16x16x32_bf16 (wave32); attention K/V chunks staged
// into LDS with double-buffered GLOBAL_LOAD_ASYNC_TO_LDS_B128 (ASYNCcnt).
// ---------------------------------------------------------------------------

typedef __bf16 bf16;
typedef __attribute__((ext_vector_type(16))) __bf16 v16bf;
typedef __attribute__((ext_vector_type(8)))  float  v8f;

#define WMMA_BF16(a, b, c) \
  __builtin_amdgcn_wmma_f32_16x16x32_bf16(false, (a), false, (b), (short)0, (c), false, false)

// A-matrix 16x32 bf16 per-lane K pattern (contiguous, matches B loaders):
// VGPR i (pair 2i,2i+1): K = 2i + 16*half
__device__ __forceinline__ int a_pat(int i, int half) {
  return 2 * i + 16 * half;
}

union U32BF { unsigned int u; bf16 h[2]; };

// A fragment from f32 row (gathered token row), converted to bf16.
__device__ __forceinline__ v16bf load_a_f32(const float* __restrict__ row, int kb, int half) {
  v16bf a;
#pragma unroll
  for (int i = 0; i < 8; ++i) {
    int k = kb + a_pat(i, half);
    a[2 * i]     = (bf16)row[k];
    a[2 * i + 1] = (bf16)row[k + 1];
  }
  return a;
}

// A fragment from bf16 row (dword pair loads; works for global or LDS rows).
__device__ __forceinline__ v16bf load_a_bf16(const bf16* row, int kb, int half) {
  v16bf a;
#pragma unroll
  for (int i = 0; i < 8; ++i) {
    int k = kb + a_pat(i, half);
    U32BF u;
    u.u = *(const unsigned int*)(row + k);
    a[2 * i]     = u.h[0];
    a[2 * i + 1] = u.h[1];
  }
  return a;
}

// B-matrix 32x16: lane holds column n = lane&15, khalf = lane>>4,
// pair 2i,2i+1 -> K = 2i + 16*khalf.  B[k][n] = row[k] where `row` is the
// source row owned by column n -> pairs contiguous in memory.
__device__ __forceinline__ v16bf load_b_f32(const float* __restrict__ row, int kb, int khalf) {
  v16bf b;
#pragma unroll
  for (int i = 0; i < 8; ++i) {
    int k = kb + 2 * i + 16 * khalf;
    b[2 * i]     = (bf16)row[k];
    b[2 * i + 1] = (bf16)row[k + 1];
  }
  return b;
}

// Same, bf16 source, K-range 0..31 (scores: B[k][n] = K[key=n][k]).
__device__ __forceinline__ v16bf load_b_bf16_contig(const bf16* row, int khalf) {
  v16bf b;
#pragma unroll
  for (int i = 0; i < 8; ++i) {
    int k = 2 * i + 16 * khalf;
    U32BF u;
    u.u = *(const unsigned int*)(row + k);
    b[2 * i]     = u.h[0];
    b[2 * i + 1] = u.h[1];
  }
  return b;
}

// V fragment: B[k][n] = tile[k][n] with row stride 32 -> scalar bf16 reads
// (from LDS after async staging; cheap ds traffic).
__device__ __forceinline__ v16bf load_b_v(const bf16* base, int n, int khalf) {
  v16bf b;
#pragma unroll
  for (int i = 0; i < 8; ++i) {
    int k = 2 * i + 16 * khalf;
    b[2 * i]     = base[k * 32 + n];
    b[2 * i + 1] = base[(k + 1) * 32 + n];
  }
  return b;
}

// Async 16B copy global -> LDS (tracked by ASYNCcnt).  LDS offset is the low
// 32 bits of the generic address.
__device__ __forceinline__ void async_copy_b128(const bf16* gsrc, bf16* ldst) {
  unsigned lds_off = (unsigned)(unsigned long long)(uintptr_t)ldst;
  unsigned long long ga = (unsigned long long)(uintptr_t)gsrc;
  asm volatile("global_load_async_to_lds_b128 %0, %1, off"
               :: "v"(lds_off), "v"(ga) : "memory");
}

// ---------------------------------------------------------------------------
// Kernel 1: gather + QKV projection.  out = x[pidx] @ W^T + b  (x3 weights)
// Wave = 16 rows x 64 cols for Q,K,V simultaneously (A fragment reused 12x).
// Output layout: [B, H, T, 32] bf16.
// ---------------------------------------------------------------------------
__global__ __launch_bounds__(256) void qkv_gemm_kernel(
    const float* __restrict__ x,
    const float* __restrict__ Wq, const float* __restrict__ bq,
    const float* __restrict__ Wk, const float* __restrict__ bk,
    const float* __restrict__ Wv, const float* __restrict__ bv,
    const int* __restrict__ pidx,
    bf16* __restrict__ qo, bf16* __restrict__ ko, bf16* __restrict__ vo) {
  const int lane = threadIdx.x & 31, half = lane >> 4, lm = lane & 15;
  const int wg = blockIdx.x * 8 + (threadIdx.x >> 5);  // 8192 wave-jobs
  const int nt = wg & 3;
  const int mtile = wg >> 2;  // 0..2047
  const int m0 = mtile * 16;
  const int n0 = nt * 64;

  const int srow = pidx[m0 + lm];
  const float* arow = x + (size_t)srow * 256;
  const float* W[3] = {Wq, Wk, Wv};

  v8f c[12] = {};
  for (int kb = 0; kb < 256; kb += 32) {
    if (kb + 32 < 256)  // global_prefetch next A chunk of the gathered row
      __builtin_prefetch(arow + kb + 32, 0, 0);
    v16bf a = load_a_f32(arow, kb, half);
#pragma unroll
    for (int mat = 0; mat < 3; ++mat) {
#pragma unroll
      for (int t = 0; t < 4; ++t) {
        const float* wrow = W[mat] + (size_t)(n0 + t * 16 + lm) * 256;
        v16bf bfrag = load_b_f32(wrow, kb, half);
        c[mat * 4 + t] = WMMA_BF16(a, bfrag, c[mat * 4 + t]);
      }
    }
  }

  const float* bias[3] = {bq, bk, bv};
  bf16* outp[3] = {qo, ko, vo};
#pragma unroll
  for (int mat = 0; mat < 3; ++mat) {
#pragma unroll
    for (int t = 0; t < 4; ++t) {
      const int n = n0 + t * 16 + lm;
      const float bl = bias[mat][n];
      const int hh = n >> 5, d = n & 31;
#pragma unroll
      for (int r = 0; r < 8; ++r) {
        const int tok = m0 + r + 8 * half;  // flat token b*512+t
        const int b_ = tok >> 9, t_ = tok & 511;
        outp[mat][((size_t)((b_ * 8 + hh) * 512 + t_)) * 32 + d] =
            (bf16)(c[mat * 4 + t][r] + bl);
      }
    }
  }
}

// ---------------------------------------------------------------------------
// Kernel 2: flash attention per (b,h).  Wave = 16 query rows; 8 waves share
// double-buffered LDS K/V tiles staged with async-to-LDS (overlaps the WMMA
// work on the previous chunk).  Scores stay in registers (online softmax);
// P is transposed C-layout -> A-layout through a wave-private LDS slab.
// Output y: [B*T, 256] bf16 (heads re-interleaved).
// ---------------------------------------------------------------------------
__global__ __launch_bounds__(256) void attn_kernel(
    const bf16* __restrict__ qb, const bf16* __restrict__ kb,
    const bf16* __restrict__ vb, const int* __restrict__ pbatch,
    bf16* __restrict__ yb) {
  __shared__ bf16 kt[2][32][32];   // [buf][key][dim]   2 x 2 KB
  __shared__ bf16 vt[2][32][32];   // [buf][key][dim]   2 x 2 KB
  __shared__ bf16 pt[8][16][32];   // per-wave P tile   8 KB

  const int lane = threadIdx.x & 31, half = lane >> 4, lm = lane & 15;
  const int wave = threadIdx.x >> 5;
  const int bh = blockIdx.x >> 2;       // 0..511
  const int rb = blockIdx.x & 3;        // 4 row-blocks of 128 queries
  const int b = bh >> 3, h = bh & 7;
  const int q0 = rb * 128 + wave * 16;

  const bf16* kbase = kb + (size_t)bh * 512 * 32;
  const bf16* vbase = vb + (size_t)bh * 512 * 32;

  // Cooperative stage of one 32-key chunk: 256 threads x 16 B = 4 KB (K+V).
  const int tid = threadIdx.x;
  const int eo = (tid & 127) * 8;  // element offset within 1024-elem tile
  auto stage = [&](int c0, int buf) {
    const bf16* src = (tid < 128 ? kbase : vbase) + (size_t)c0 * 32 + eo;
    bf16* dst = (tid < 128 ? &kt[buf][0][0] : &vt[buf][0][0]) + eo;
    async_copy_b128(src, dst);
  };

  const bf16* qrow = qb + ((size_t)bh * 512 + q0 + lm) * 32;
  const v16bf qa = load_a_bf16(qrow, 0, half);

  const int* pb = pbatch + b * 512;
  int qbid[8];
#pragma unroll
  for (int r = 0; r < 8; ++r) qbid[r] = pb[q0 + r + 8 * half];

  float m_[8], l_[8];
#pragma unroll
  for (int r = 0; r < 8; ++r) { m_[r] = -3.0e38f; l_[r] = 0.0f; }
  v8f o0 = {}, o1 = {};

  const float scale = 0.17677669529663687f;  // 1/sqrt(32)

  stage(0, 0);
  asm volatile("s_wait_asynccnt 0" ::: "memory");
  __syncthreads();

  for (int c0 = 0; c0 < 512; c0 += 32) {
    const int buf = (c0 >> 5) & 1;
    if (c0 + 32 < 512) stage(c0 + 32, buf ^ 1);  // overlap with compute below

    // ---- scores: two 16x16 tiles from the LDS K tile -------------------
    v8f s0t, s1t;
    {
      const v8f z = {};
      v16bf kf0 = load_b_bf16_contig(&kt[buf][lm][0], half);
      s0t = WMMA_BF16(qa, kf0, z);
      v16bf kf1 = load_b_bf16_contig(&kt[buf][16 + lm][0], half);
      s1t = WMMA_BF16(qa, kf1, z);
    }
    const int kbid0 = pb[c0 + lm];
    const int kbid1 = pb[c0 + 16 + lm];

    // ---- online softmax update (row reductions within 16-lane half) ----
    float p0[8], p1[8];
#pragma unroll
    for (int r = 0; r < 8; ++r) {
      float s0 = s0t[r] * scale;
      if (qbid[r] == kbid0) s0 = -1.0e9f;   // reference masks where equal
      float s1 = s1t[r] * scale;
      if (qbid[r] == kbid1) s1 = -1.0e9f;
      float mx = fmaxf(s0, s1);
      mx = fmaxf(mx, __shfl_xor(mx, 1, 32));
      mx = fmaxf(mx, __shfl_xor(mx, 2, 32));
      mx = fmaxf(mx, __shfl_xor(mx, 4, 32));
      mx = fmaxf(mx, __shfl_xor(mx, 8, 32));
      const float mn = fmaxf(m_[r], mx);
      const float al = __expf(m_[r] - mn);
      const float e0 = __expf(s0 - mn);
      const float e1 = __expf(s1 - mn);
      float rs = e0 + e1;
      rs += __shfl_xor(rs, 1, 32);
      rs += __shfl_xor(rs, 2, 32);
      rs += __shfl_xor(rs, 4, 32);
      rs += __shfl_xor(rs, 8, 32);
      l_[r] = l_[r] * al + rs;
      m_[r] = mn;
      o0[r] *= al;
      o1[r] *= al;
      p0[r] = e0;
      p1[r] = e1;
    }

    // ---- transpose P (C-layout -> A-layout) via wave-private LDS -------
#pragma unroll
    for (int r = 0; r < 8; ++r) {
      pt[wave][r + 8 * half][lm]      = (bf16)p0[r];
      pt[wave][r + 8 * half][16 + lm] = (bf16)p1[r];
    }
    asm volatile("s_wait_dscnt 0" ::: "memory");
    const v16bf pf = load_a_bf16(&pt[wave][lm][0], 0, half);

    // ---- O += P @ V (two head-dim halves) from the LDS V tile ----------
    v16bf v0 = load_b_v(&vt[buf][0][0], lm, half);
    o0 = WMMA_BF16(pf, v0, o0);
    v16bf v1 = load_b_v(&vt[buf][0][0], 16 + lm, half);
    o1 = WMMA_BF16(pf, v1, o1);

    // next chunk's staging must be complete & visible to all waves
    asm volatile("s_wait_asynccnt 0" ::: "memory");
    __syncthreads();
  }

  // ---- finalize and write y in [B*T, 256] bf16 layout ------------------
#pragma unroll
  for (int r = 0; r < 8; ++r) {
    const float inv = 1.0f / l_[r];
    const size_t tok = (size_t)b * 512 + q0 + r + 8 * half;
    yb[tok * 256 + h * 32 + lm]      = (bf16)(o0[r] * inv);
    yb[tok * 256 + h * 32 + 16 + lm] = (bf16)(o1[r] * inv);
  }
}

// ---------------------------------------------------------------------------
// Kernel 3: output projection with fused inverse-index gather:
//   out[i,:] = y[pinv[i],:] @ Wp^T + bp     (out f32, 24576 x 256)
// ---------------------------------------------------------------------------
__global__ __launch_bounds__(256) void out_proj_kernel(
    const bf16* __restrict__ yb, const float* __restrict__ Wp,
    const float* __restrict__ bp, const int* __restrict__ pinv,
    float* __restrict__ out) {
  const int lane = threadIdx.x & 31, half = lane >> 4, lm = lane & 15;
  const int wg = blockIdx.x * 8 + (threadIdx.x >> 5);  // 6144 wave-jobs
  const int nt = wg & 3;
  const int mtile = wg >> 2;  // 0..1535
  const int m0 = mtile * 16;
  const int n0 = nt * 64;

  const int src = pinv[m0 + lm];
  const bf16* arow = yb + (size_t)src * 256;

  v8f c[4] = {};
  for (int kb = 0; kb < 256; kb += 32) {
    v16bf a = load_a_bf16(arow, kb, half);
#pragma unroll
    for (int t = 0; t < 4; ++t) {
      const float* wrow = Wp + (size_t)(n0 + t * 16 + lm) * 256;
      v16bf bfrag = load_b_f32(wrow, kb, half);
      c[t] = WMMA_BF16(a, bfrag, c[t]);
    }
  }

#pragma unroll
  for (int t = 0; t < 4; ++t) {
    const int n = n0 + t * 16 + lm;
    const float bl = bp[n];
#pragma unroll
    for (int r = 0; r < 8; ++r) {
      const size_t row = (size_t)(m0 + r + 8 * half);
      out[row * 256 + n] = c[t][r] + bl;
    }
  }
}

// ---------------------------------------------------------------------------
extern "C" void kernel_launch(void* const* d_in, const int* in_sizes, int n_in,
                              void* d_out, int out_size, void* d_ws,
                              size_t ws_size, hipStream_t stream) {
  (void)in_sizes; (void)n_in; (void)out_size; (void)ws_size;

  const float* x  = (const float*)d_in[0];
  const float* Wq = (const float*)d_in[1];
  const float* bq = (const float*)d_in[2];
  const float* Wk = (const float*)d_in[3];
  const float* bk = (const float*)d_in[4];
  const float* Wv = (const float*)d_in[5];
  const float* bv = (const float*)d_in[6];
  const float* Wp = (const float*)d_in[7];
  const float* bp = (const float*)d_in[8];
  const int* pidx   = (const int*)d_in[9];
  const int* pbatch = (const int*)d_in[10];
  const int* pinv   = (const int*)d_in[11];
  float* out = (float*)d_out;

  // Workspace: Q,K,V ([B,H,T,32] bf16) + y ([B*T,256] bf16) = 4 x 16 MB.
  const size_t NELEM = (size_t)32768 * 256;
  bf16* qb = (bf16*)d_ws;
  bf16* kbuf = qb + NELEM;
  bf16* vbuf = kbuf + NELEM;
  bf16* ybuf = vbuf + NELEM;

  // 1) gather + QKV: 8192 wave-jobs -> 1024 blocks x 256 threads
  qkv_gemm_kernel<<<1024, 256, 0, stream>>>(x, Wq, bq, Wk, bk, Wv, bv, pidx,
                                            qb, kbuf, vbuf);
  // 2) flash attention: 512 (b,h) x 4 row-blocks -> 2048 blocks
  attn_kernel<<<2048, 256, 0, stream>>>(qb, kbuf, vbuf, pbatch, ybuf);
  // 3) projection + scatter: 6144 wave-jobs -> 768 blocks
  out_proj_kernel<<<768, 256, 0, stream>>>(ybuf, Wp, bp, pinv, out);
}
